// RKDAngleLoss_86723979641277
// MI455X (gfx1250) — matrix operand order repaired and checked by reading, not compiled, not run on hardware
//
#include <hip/hip_runtime.h>
#include <hip/hip_bf16.h>

// ---------------------------------------------------------------------------
// RKD angle loss on MI455X (gfx1250, wave32).
//   s = student@W1+b1, t = teacher@W2+b2            (512x128 each, f32)
//   For each anchor i: N_i[j] = normalize(x[j]-x[i]); G_i = N_i N_i^T
//   loss = mean |Gs - Gt| over 512^3 elements.
// Heavy part: 1024 GEMMs (512x128)@(128x512) = ~68.7 GFLOP, L2-resident
// (512 KB working set vs 192 MB L2) -> matrix-core bound. Use
// v_wmma_f32_16x16x32_f16 (f16 A/B of unit-normalized values, f32 accum).
// ---------------------------------------------------------------------------

typedef __attribute__((ext_vector_type(16))) _Float16 v16h;
typedef __attribute__((ext_vector_type(8)))  _Float16 v8h;
typedef __attribute__((ext_vector_type(4)))  _Float16 v4h;
typedef __attribute__((ext_vector_type(8)))  float    v8f;
typedef __attribute__((ext_vector_type(4)))  float    v4f;

#define N_PTS   512
#define D_IN    512
#define D_EMB   128
#define TILE    64
#define NTJ     (N_PTS / TILE)          // 8 tiles per dim
#define NBLK    (N_PTS * NTJ * NTJ)     // 32768 blocks
#define LSTR    136                     // LDS row stride in halves (272B: bank-conflict-free)
#define EPS_T   1e-12f

// ---------------------------------------------------------------------------
// Kernel 1: embedding GEMM  out[n][c] = sum_k X[n][k]*W[k][c] + b[c]
// 67 MFLOP per call -> simple VALU kernel is plenty.
// ---------------------------------------------------------------------------
__global__ __launch_bounds__(256) void rkd_embed_kernel(
    const float* __restrict__ X, const float* __restrict__ W,
    const float* __restrict__ b, float* __restrict__ out) {
  const int idx = blockIdx.x * blockDim.x + threadIdx.x;  // 0..65535
  const int n = idx >> 7;
  const int c = idx & (D_EMB - 1);
  float acc = b[c];
  const float* xr = X + n * D_IN;
#pragma unroll 8
  for (int k = 0; k < D_IN; ++k)
    acc = fmaf(xr[k], W[k * D_EMB + c], acc);
  out[idx] = acc;
}

// ---------------------------------------------------------------------------
// Kernel 2: per-anchor normalized-difference Gram tiles + |Gs-Gt| partial sums
// grid = (kt=8, jt=8, anchor=512), block = 256 threads = 8 wave32s.
// ---------------------------------------------------------------------------
__device__ __forceinline__ void stage_rows(const float* __restrict__ emb,
                                           int anchor, int jBase, int kBase,
                                           float* ancLDS, _Float16* Aj, _Float16* Ak) {
  const int t = threadIdx.x;
  // anchor row -> LDS (f32)
  if (t < D_EMB) ancLDS[t] = emb[anchor * D_EMB + t];
  __syncthreads();

  // 2 threads per row: 2 arrays x 64 rows x 2 half-rows = 256 threads
  const int which = t >> 7;            // 0 = Aj (j rows), 1 = Ak (k rows)
  const int r     = (t & 127) >> 1;    // row in tile, 0..63
  const int hh    = t & 1;             // column half: 0 -> 0..63, 1 -> 64..127
  const int gRow  = (which ? kBase : jBase) + r;
  const float* __restrict__ src = emb + gRow * D_EMB + hh * 64;
  const float* __restrict__ anc = ancLDS + hh * 64;
  _Float16* dst = (which ? Ak : Aj) + r * LSTR + hh * 64;

  float ss = 0.f;
#pragma unroll
  for (int c4 = 0; c4 < 16; ++c4) {
    v4f d = ((const v4f*)src)[c4] - ((const v4f*)anc)[c4];
    ss += d.x * d.x + d.y * d.y + d.z * d.z + d.w * d.w;
  }
  ss += __shfl_xor(ss, 1, 32);                       // pair lanes share a row
  const float scale = 1.0f / fmaxf(sqrtf(ss), EPS_T); // torch F.normalize eps

#pragma unroll
  for (int c8 = 0; c8 < 8; ++c8) {
    v4f d0 = (((const v4f*)src)[2 * c8]     - ((const v4f*)anc)[2 * c8])     * scale;
    v4f d1 = (((const v4f*)src)[2 * c8 + 1] - ((const v4f*)anc)[2 * c8 + 1]) * scale;
    v4h h0 = __builtin_convertvector(d0, v4h);
    v4h h1 = __builtin_convertvector(d1, v4h);
    *(v8h*)(dst + 8 * c8) = __builtin_shufflevector(h0, h1, 0,1,2,3,4,5,6,7);
  }
  __syncthreads();
}

__device__ __forceinline__ void gram_accum(const _Float16* __restrict__ Aj,
                                           const _Float16* __restrict__ Ak,
                                           int waveId, int lane, v8f acc[2]) {
  const int m  = lane & 15;
  const int hi = lane >> 4;
#pragma unroll
  for (int tt = 0; tt < 2; ++tt) {
    const int tileIdx = waveId * 2 + tt;             // 16 tiles over 8 waves
    const int mt = tileIdx >> 2, nt = tileIdx & 3;
    const _Float16* __restrict__ arow = Aj + (mt * 16 + m) * LSTR;
    const _Float16* __restrict__ brow = Ak + (nt * 16 + m) * LSTR;
    v8f c = acc[tt];
#pragma unroll
    for (int kk = 0; kk < D_EMB; kk += 32) {
      // A 16x32 f16 layout (wave32): elems 0..7 = K kk+hi*8+{0..7},
      //                              elems 8..15 = K kk+16+hi*8+{0..7}
      v8h a0 = *(const v8h*)(arow + kk + hi * 8);
      v8h a1 = *(const v8h*)(arow + kk + 16 + hi * 8);
      // B 32x16 f16 layout: elems 0..7 = K kk+hi*8+{0..7}, elems 8..15 = K kk+16+hi*8+{0..7} of column n=m
      v8h b0 = *(const v8h*)(brow + kk + hi * 8);
      v8h b1 = *(const v8h*)(brow + kk + 16 + hi * 8);
      v16h a = __builtin_shufflevector(a0, a1, 0,1,2,3,4,5,6,7,8,9,10,11,12,13,14,15);
      v16h b = __builtin_shufflevector(b0, b1, 0,1,2,3,4,5,6,7,8,9,10,11,12,13,14,15);
      c = __builtin_amdgcn_wmma_f32_16x16x32_f16(false, a, false, b,
                                                 (short)0, c, false, false);
    }
    acc[tt] = c;
  }
}

__global__ __launch_bounds__(256) void rkd_angle_tile_kernel(
    const float* __restrict__ sEmb, const float* __restrict__ tEmb,
    float* __restrict__ partials) {
  __shared__ _Float16 Aj[TILE * LSTR];
  __shared__ _Float16 Ak[TILE * LSTR];
  __shared__ float ancLDS[D_EMB];
  __shared__ float wsum[8];

  const int t      = threadIdx.x;
  const int lane   = t & 31;
  const int waveId = t >> 5;
  const int anchor = blockIdx.z;
  const int jBase  = blockIdx.y * TILE;
  const int kBase  = blockIdx.x * TILE;

  v8f accS[2] = {(v8f)(0.f), (v8f)(0.f)};
  v8f accT[2] = {(v8f)(0.f), (v8f)(0.f)};

  // Student pass
  stage_rows(sEmb, anchor, jBase, kBase, ancLDS, Aj, Ak);
  gram_accum(Aj, Ak, waveId, lane, accS);
  __syncthreads();                       // all waves done reading LDS
  // Teacher pass (restage same LDS)
  stage_rows(tEmb, anchor, jBase, kBase, ancLDS, Aj, Ak);
  gram_accum(Aj, Ak, waveId, lane, accT);

  // |Gs - Gt| local sum
  float local = 0.f;
#pragma unroll
  for (int tt = 0; tt < 2; ++tt)
#pragma unroll
    for (int v = 0; v < 8; ++v)
      local += fabsf(accS[tt][v] - accT[tt][v]);

#pragma unroll
  for (int off = 16; off > 0; off >>= 1)
    local += __shfl_xor(local, off, 32);
  if (lane == 0) wsum[waveId] = local;
  __syncthreads();
  if (t == 0) {
    float s = 0.f;
#pragma unroll
    for (int w = 0; w < 8; ++w) s += wsum[w];
    partials[(blockIdx.z * NTJ + blockIdx.y) * NTJ + blockIdx.x] = s;
  }
}

// ---------------------------------------------------------------------------
// Kernel 3: deterministic single-block reduction of 32768 partials -> mean
// ---------------------------------------------------------------------------
__global__ __launch_bounds__(256) void rkd_reduce_kernel(
    const float* __restrict__ partials, float* __restrict__ out) {
  __shared__ float sm[256];
  const int t = threadIdx.x;
  float local = 0.f;
  for (int i = t; i < NBLK; i += 256) local += partials[i];
  sm[t] = local;
  __syncthreads();
#pragma unroll
  for (int s = 128; s > 0; s >>= 1) {
    if (t < s) sm[t] += sm[t + s];
    __syncthreads();
  }
  if (t == 0)
    out[0] = sm[0] * (1.0f / ((float)N_PTS * (float)N_PTS * (float)N_PTS));
}

extern "C" void kernel_launch(void* const* d_in, const int* in_sizes, int n_in,
                              void* d_out, int out_size, void* d_ws, size_t ws_size,
                              hipStream_t stream) {
  (void)in_sizes; (void)n_in; (void)out_size; (void)ws_size;
  const float* student = (const float*)d_in[0];
  const float* teacher = (const float*)d_in[1];
  const float* W1      = (const float*)d_in[2];
  const float* b1      = (const float*)d_in[3];
  const float* W2      = (const float*)d_in[4];
  const float* b2      = (const float*)d_in[5];
  float* out = (float*)d_out;

  // workspace layout: sEmb (256KB) | tEmb (256KB) | partials (128KB)
  float* sEmb     = (float*)d_ws;
  float* tEmb     = sEmb + N_PTS * D_EMB;
  float* partials = tEmb + N_PTS * D_EMB;

  // embeddings
  rkd_embed_kernel<<<dim3((N_PTS * D_EMB) / 256), 256, 0, stream>>>(student, W1, b1, sEmb);
  rkd_embed_kernel<<<dim3((N_PTS * D_EMB) / 256), 256, 0, stream>>>(teacher, W2, b2, tEmb);

  // per-anchor Gram-difference tiles (WMMA)
  rkd_angle_tile_kernel<<<dim3(NTJ, NTJ, N_PTS), 256, 0, stream>>>(sEmb, tEmb, partials);

  // final deterministic reduction
  rkd_reduce_kernel<<<dim3(1), 256, 0, stream>>>(partials, out);
}
